// AdaptiveTopKChannelStack_13073880449229
// MI455X (gfx1250) — matrix operand order repaired and checked by reading, not compiled, not run on hardware
//
#include <hip/hip_runtime.h>
#include <stdint.h>

// ---------------------------------------------------------------------------
// Problem constants (from reference)
// ---------------------------------------------------------------------------
#define B_ROWS 32768
#define D_DIM  512
#define E_COMP 16
#define CH_DIM 128
#define N_DIM  (E_COMP * CH_DIM)   // 2048

typedef __bf16 bf16_t;
typedef bf16_t v16bf __attribute__((ext_vector_type(16)));
typedef float  v8f   __attribute__((ext_vector_type(8)));

// ---------------------------------------------------------------------------
// bf16 split helpers (round-to-nearest-even)
// ---------------------------------------------------------------------------
static __device__ __forceinline__ uint32_t f32_to_bf16_bits(float f) {
    uint32_t u = __float_as_uint(f);
    u = u + 0x7FFFu + ((u >> 16) & 1u);
    return u >> 16;
}

// hi = bf16(f) (RNE), lo = f - float(hi)
static __device__ __forceinline__ void split_bf16(float f, uint32_t& hi, float& lo) {
    uint32_t u = __float_as_uint(f);
    uint32_t r = u + 0x7FFFu + ((u >> 16) & 1u);
    hi = r >> 16;
    lo = f - __uint_as_float(r & 0xFFFF0000u);
}

// ---------------------------------------------------------------------------
// Kernel 1: noisy top-1 gate -> k[b] = argmax_e(H[b,e])
// 16 threads per row (one per expert), shfl_xor argmax in 16-lane groups.
// ---------------------------------------------------------------------------
__global__ __launch_bounds__(256)
void gate_argmax_kernel(const float* __restrict__ x,
                        const float* __restrict__ Wg_w,
                        const float* __restrict__ Wg_b,
                        const float* __restrict__ Wn_w,
                        const float* __restrict__ Wn_b,
                        const float* __restrict__ eps,
                        int* __restrict__ ksel)
{
    const int tid = threadIdx.x;
    const int rl  = tid >> 4;          // local row 0..15
    const int e   = tid & 15;          // expert 0..15
    const int row = blockIdx.x * 16 + rl;

    const float* __restrict__ xr = x + (size_t)row * D_DIM;

    float ag = Wg_b[e];
    float an = Wn_b[e];
#pragma unroll 4
    for (int d = 0; d < D_DIM; ++d) {
        const float xv = xr[d];
        ag = fmaf(xv, Wg_w[d * E_COMP + e], ag);
        an = fmaf(xv, Wn_w[d * E_COMP + e], an);
    }
    // softplus(an) = log1p(exp(an)), overflow-safe
    const float sp = (an > 20.0f) ? an : log1pf(__expf(an));
    float best = ag + eps[e] * sp;
    int   bidx = e;

    // argmax over the 16 lanes of this row (first max wins, like jnp.argmax)
#pragma unroll
    for (int off = 8; off > 0; off >>= 1) {
        const float ov = __shfl_xor(best, off, 16);
        const int   oi = __shfl_xor(bidx, off, 16);
        if (ov > best || (ov == best && oi < bidx)) { best = ov; bidx = oi; }
    }
    if (e == 0) ksel[row] = bidx;
}

// ---------------------------------------------------------------------------
// Kernel 2: masked expert GEMM via bf16 hi/lo split WMMA.
// Block = 256 threads (8 waves) computes a 128-row x 128-col tile.
// blockIdx.x = expert e (N tile), blockIdx.y = row tile.
// ---------------------------------------------------------------------------
__global__ __launch_bounds__(256)
void moe_gemm_wmma_kernel(const float* __restrict__ x,
                          const float* __restrict__ Wc,
                          const float* __restrict__ bc,
                          const int*   __restrict__ ksel,
                          float*       __restrict__ out)
{
    // LDS tiles stored directly in WMMA fragment order.
    // Each entry is a packed bf16 pair (K, K+1) -> (bits[15:0], bits[31:16]).
    __shared__ __attribute__((aligned(32))) uint32_t aHiLds[2][2048];
    __shared__ __attribute__((aligned(32))) uint32_t aLoLds[2][2048];
    __shared__ __attribute__((aligned(32))) uint32_t bHiLds[2][2048];
    __shared__ __attribute__((aligned(32))) uint32_t bLoLds[2][2048];

    const int tid = threadIdx.x;
    const int e   = blockIdx.x;
    const int m0  = blockIdx.y * 128;
    const float* __restrict__ wbase = Wc + (size_t)e * (D_DIM * CH_DIM);

    // ---------------- producer index precompute -----------------------------
    // A tile: 128 rows x 32 K fp32 -> 2048 bf16-pairs. 8 pairs / thread.
    // fragment layout (ISA 16-bit A 16x32): lane L = h*16 + m,
    //   VGPR r: K = 16*(r>>2) + 8*h + 2*(r&3) + {0,1}
    int aOff[8], aIdx[8];
    int bOff[8], bIdx[8];
#pragma unroll
    for (int i = 0; i < 8; ++i) {
        const int p    = tid + i * 256;        // 0..2047
        const int rowL = p >> 4;               // 0..127
        const int k2   = p & 15;               // pair index within K-chunk
        const int h    = (k2 >> 2) & 1;
        const int r    = ((k2 >> 3) << 2) | (k2 & 3);
        const int rb   = rowL >> 4;
        const int m    = rowL & 15;
        const int L    = h * 16 + m;
        aIdx[i] = (rb * 32 + L) * 8 + r;
        aOff[i] = (m0 + rowL) * D_DIM + 2 * k2;   // + kt*32 at load time
    }
    // B tile: 32 K x 128 cols -> 2048 pairs. Pair (K,K+1) strides CH in memory.
    // fragment layout (B 32x16 bf16): lane L = h*16 + n, VGPR r: K = 16*(r>>2) + 8*h + 2*(r&3)
#pragma unroll
    for (int i = 0; i < 8; ++i) {
        const int p   = tid + i * 256;
        const int col = p & 127;
        const int k2  = p >> 7;                // 0..15
        const int h   = (k2 >> 2) & 1;
        const int r   = ((k2 >> 3) << 2) | (k2 & 3);
        const int cb  = col >> 4;
        const int L   = h * 16 + (col & 15);
        bIdx[i] = (cb * 32 + L) * 8 + r;
        bOff[i] = (2 * k2) * CH_DIM + col;     // + kt*32*CH at load time
    }

    // ---------------- accumulators ------------------------------------------
    const int lane = tid & 31;
    const int wm   = (tid >> 5) & 3;   // row-block pair owner (rb = wm*2 + ai)
    const int wn   = tid >> 7;         // col-quad owner (cb = wn*4 + j)

    v8f acc[2][4];
    const v8f vzero = {0.f, 0.f, 0.f, 0.f, 0.f, 0.f, 0.f, 0.f};
#pragma unroll
    for (int ai = 0; ai < 2; ++ai)
#pragma unroll
        for (int j = 0; j < 4; ++j) acc[ai][j] = vzero;

    // ---------------- pipelined K loop --------------------------------------
    float2 aval[8];
    float  bv0[8], bv1[8];

    // prefetch chunk 0
#pragma unroll
    for (int i = 0; i < 8; ++i) aval[i] = *(const float2*)(x + aOff[i]);
#pragma unroll
    for (int i = 0; i < 8; ++i) {
        bv0[i] = wbase[bOff[i]];
        bv1[i] = wbase[bOff[i] + CH_DIM];
    }

    int buf = 0;
    for (int kt = 0; kt < D_DIM / 32; ++kt) {
        // stage current chunk into LDS (split into bf16 hi/lo pairs)
#pragma unroll
        for (int i = 0; i < 8; ++i) {
            uint32_t h0, h1; float l0, l1;
            split_bf16(aval[i].x, h0, l0);
            split_bf16(aval[i].y, h1, l1);
            aHiLds[buf][aIdx[i]] = h0 | (h1 << 16);
            aLoLds[buf][aIdx[i]] = f32_to_bf16_bits(l0) | (f32_to_bf16_bits(l1) << 16);
            uint32_t g0, g1; float m0f, m1f;
            split_bf16(bv0[i], g0, m0f);
            split_bf16(bv1[i], g1, m1f);
            bHiLds[buf][bIdx[i]] = g0 | (g1 << 16);
            bLoLds[buf][bIdx[i]] = f32_to_bf16_bits(m0f) | (f32_to_bf16_bits(m1f) << 16);
        }
        __syncthreads();

        // prefetch next chunk while WMMAs run
        if (kt + 1 < D_DIM / 32) {
            const int ka = (kt + 1) * 32;
            const int kb = (kt + 1) * 32 * CH_DIM;
#pragma unroll
            for (int i = 0; i < 8; ++i) aval[i] = *(const float2*)(x + aOff[i] + ka);
#pragma unroll
            for (int i = 0; i < 8; ++i) {
                bv0[i] = wbase[bOff[i] + kb];
                bv1[i] = wbase[bOff[i] + kb + CH_DIM];
            }
        }

        // A fragments for this wave's two row-blocks
        v16bf aH[2], aL[2];
#pragma unroll
        for (int ai = 0; ai < 2; ++ai) {
            const int rb = wm * 2 + ai;
            aH[ai] = *(const v16bf*)&aHiLds[buf][(rb * 32 + lane) * 8];
            aL[ai] = *(const v16bf*)&aLoLds[buf][(rb * 32 + lane) * 8];
        }
        // loop over this wave's four col-blocks; 3 split-WMMAs each
#pragma unroll
        for (int j = 0; j < 4; ++j) {
            const int cb = wn * 4 + j;
            const v16bf bH = *(const v16bf*)&bHiLds[buf][(cb * 32 + lane) * 8];
            const v16bf bL = *(const v16bf*)&bLoLds[buf][(cb * 32 + lane) * 8];
#pragma unroll
            for (int ai = 0; ai < 2; ++ai) {
                acc[ai][j] = __builtin_amdgcn_wmma_f32_16x16x32_bf16(
                    false, aH[ai], false, bH, (short)0, acc[ai][j], false, false);
                acc[ai][j] = __builtin_amdgcn_wmma_f32_16x16x32_bf16(
                    false, aH[ai], false, bL, (short)0, acc[ai][j], false, false);
                acc[ai][j] = __builtin_amdgcn_wmma_f32_16x16x32_bf16(
                    false, aL[ai], false, bH, (short)0, acc[ai][j], false, false);
            }
        }
        __syncthreads();   // safe reuse of this buffer two iterations later
        buf ^= 1;
    }

    // ---------------- epilogue: +bias, monotonic top-1 mask, store ----------
    // C/D fragment layout: lane L -> col = L%16, VGPR v -> row = v + 8*(L/16)
    const int colIn   = lane & 15;
    const int rowHalf = (lane >> 4) * 8;
#pragma unroll
    for (int ai = 0; ai < 2; ++ai) {
        const int rb    = wm * 2 + ai;
        const int rbase = m0 + rb * 16 + rowHalf;
        int kk[8];
#pragma unroll
        for (int v = 0; v < 8; ++v) kk[v] = ksel[rbase + v];
#pragma unroll
        for (int j = 0; j < 4; ++j) {
            const int cb   = wn * 4 + j;
            const int cloc = cb * 16 + colIn;          // 0..127 within expert
            const int gcol = e * CH_DIM + cloc;        // global output column
            const float bias = bc[e * CH_DIM + cloc];
#pragma unroll
            for (int v = 0; v < 8; ++v) {
                const float val = acc[ai][j][v] + bias;
                out[(size_t)(rbase + v) * N_DIM + gcol] = (e <= kk[v]) ? val : 0.0f;
            }
        }
    }
}

// ---------------------------------------------------------------------------
// Host-side launch
// ---------------------------------------------------------------------------
extern "C" void kernel_launch(void* const* d_in, const int* in_sizes, int n_in,
                              void* d_out, int out_size, void* d_ws, size_t ws_size,
                              hipStream_t stream)
{
    const float* x     = (const float*)d_in[0];
    const float* Wc    = (const float*)d_in[1];
    const float* bc    = (const float*)d_in[2];
    const float* Wg_w  = (const float*)d_in[3];
    const float* Wg_b  = (const float*)d_in[4];
    const float* Wn_w  = (const float*)d_in[5];
    const float* Wn_b  = (const float*)d_in[6];
    const float* eps   = (const float*)d_in[7];
    float*       out   = (float*)d_out;
    int*         ksel  = (int*)d_ws;                   // 32768 ints = 128 KB

    gate_argmax_kernel<<<B_ROWS / 16, 256, 0, stream>>>(
        x, Wg_w, Wg_b, Wn_w, Wn_b, eps, ksel);

    dim3 grid(E_COMP, B_ROWS / 128);                   // 16 x 256 = 4096 blocks
    moe_gemm_wmma_kernel<<<grid, 256, 0, stream>>>(x, Wc, bc, ksel, out);
}
